// MultiHeadSelfAttention_89249420411509
// MI455X (gfx1250) — compile-verified, matrix operand order checked by reading, not run on hardware
//
#include <hip/hip_runtime.h>

// MI455X (gfx1250) multi-head self-attention, WMMA f16 path, f32 accumulate.
// Data movement: QKV GEMM emits f16 (Q,K row-major; V transposed), attention
// streams K/V^T chunks into LDS with global_load_async_to_lds_b128 (ASYNCcnt).
// Workspace layout (needs >= 37,748,736 bytes):
//   qkvh : [4096][2304] f16   (Q | K | V-region-unused)
//   vT   : [768][4096]  f16   (V transposed: [h*64+d][key])
//   attn : [4096][768]  f32   (pre-output-projection context)

#define D_MODEL 768
#define NHEAD   12
#define DH      64
#define SEQ     4096

typedef __attribute__((ext_vector_type(16))) _Float16 v16h;
typedef __attribute__((ext_vector_type(8)))  float    v8f;

union Frag { v16h v; unsigned int u[8]; };
union PK8  { _Float16 h[8]; uint4 q; };

// ISA 7.12.2: 16-bit A-matrix 16x32. Lane group g = lane/16.
__device__ __forceinline__ int kstart(int j, int g) {
    return (j < 4 ? 2 * j : 16 + 2 * (j - 4)) + 8 * g;
}

__device__ __forceinline__ unsigned int pkh2(float a, float b) {
    union { _Float16 h[2]; unsigned int u; } p;
    p.h[0] = (_Float16)a; p.h[1] = (_Float16)b;
    return p.u;
}

__device__ __forceinline__ v8f wmma_f16(const Frag& a, const Frag& b, v8f c) {
    return __builtin_amdgcn_wmma_f32_16x16x32_f16(
        false, a.v, false, b.v, (short)0, c, false, false);
}

// ---------------------------------------------------------------------------
// QKV projection: qkvh/vT = f16(x @ w_qkv + b_qkv)
// Block 64x64, 8 waves; wave -> 16x32 C sub-tile (2 accumulators).
// ---------------------------------------------------------------------------
__global__ __launch_bounds__(256)
void gemm_qkv_wmma(const float* __restrict__ A, const float* __restrict__ B,
                   const float* __restrict__ bias,
                   _Float16* __restrict__ qkvh, _Float16* __restrict__ vT)
{
    const int N = 3 * D_MODEL, K = D_MODEL;
    __shared__ _Float16 Asub[64][40];   // [m][k], 80B rows (16B aligned)
    __shared__ _Float16 Bsub[64][40];   // transposed: [n][k]

    const int tid  = threadIdx.x;
    const int lane = tid & 31;
    const int wid  = tid >> 5;
    const int g    = lane >> 4;
    const int ln   = lane & 15;

    const int blockM = blockIdx.y * 64;
    const int blockN = blockIdx.x * 64;
    const int waveM  = (wid >> 1) * 16;
    const int waveN  = (wid & 1) * 32;

    v8f acc0 = {}, acc1 = {};

    const int rowA = tid >> 2, kbA = (tid & 3) * 8;
    const int kB   = tid >> 3, nbB = (tid & 7) * 8;

    for (int k0 = 0; k0 < K; k0 += 32) {
        {   // stage A (f32 -> f16)
            const float4* p = (const float4*)(A + (size_t)(blockM + rowA) * K + k0 + kbA);
            float4 a = p[0], b = p[1];
            PK8 pk;
            pk.h[0] = (_Float16)a.x; pk.h[1] = (_Float16)a.y;
            pk.h[2] = (_Float16)a.z; pk.h[3] = (_Float16)a.w;
            pk.h[4] = (_Float16)b.x; pk.h[5] = (_Float16)b.y;
            pk.h[6] = (_Float16)b.z; pk.h[7] = (_Float16)b.w;
            *(uint4*)&Asub[rowA][kbA] = pk.q;
            if (k0 + 32 < K) __builtin_prefetch((const char*)p + 128, 0, 1);
        }
        {   // stage B transposed
            const float4* p = (const float4*)(B + (size_t)(k0 + kB) * N + blockN + nbB);
            float4 a = p[0], b = p[1];
            _Float16 h[8] = {(_Float16)a.x, (_Float16)a.y, (_Float16)a.z, (_Float16)a.w,
                             (_Float16)b.x, (_Float16)b.y, (_Float16)b.z, (_Float16)b.w};
            #pragma unroll
            for (int i = 0; i < 8; ++i) Bsub[nbB + i][kB] = h[i];
            if (k0 + 32 < K) __builtin_prefetch((const char*)p + (size_t)32 * N * 4, 0, 1);
        }
        __syncthreads();

        Frag fa, fb0, fb1;
        #pragma unroll
        for (int j = 0; j < 8; ++j) {
            const int ks = kstart(j, g);
            fa.u[j]  = *(const unsigned int*)&Asub[waveM + ln][ks];
            fb0.u[j] = *(const unsigned int*)&Bsub[waveN + ln][ks];
            fb1.u[j] = *(const unsigned int*)&Bsub[waveN + 16 + ln][ks];
        }
        acc0 = wmma_f16(fa, fb0, acc0);
        acc1 = wmma_f16(fa, fb1, acc1);
        __syncthreads();
    }

    // epilogue: Q/K region -> qkvh row-major f16; V region -> vT transposed f16
    const int rbase = blockM + waveM + 8 * g;
    #pragma unroll
    for (int t = 0; t < 2; ++t) {
        const int col = blockN + waveN + t * 16 + ln;   // region uniform per tile
        const float bv = bias[col];
        const v8f& acc = t ? acc1 : acc0;
        if (col < 2 * D_MODEL) {
            #pragma unroll
            for (int r = 0; r < 8; ++r)
                qkvh[(size_t)(rbase + r) * N + col] = (_Float16)(acc[r] + bv);
        } else {
            _Float16* vp = vT + (size_t)(col - 2 * D_MODEL) * SEQ + rbase;
            #pragma unroll
            for (int r = 0; r < 4; ++r)
                *(unsigned int*)(vp + 2 * r) = pkh2(acc[2 * r] + bv, acc[2 * r + 1] + bv);
        }
    }
}

// ---------------------------------------------------------------------------
// Output projection: C[M,N] = A[M,K] @ B[K,N] + bias  (f32 in/out)
// ---------------------------------------------------------------------------
__global__ __launch_bounds__(256)
void gemm_bias_wmma(const float* __restrict__ A, const float* __restrict__ B,
                    const float* __restrict__ bias, float* __restrict__ C,
                    int M, int N, int K)
{
    __shared__ _Float16 Asub[64][40];
    __shared__ _Float16 Bsub[64][40];

    const int tid  = threadIdx.x;
    const int lane = tid & 31;
    const int wid  = tid >> 5;
    const int g    = lane >> 4;
    const int ln   = lane & 15;

    const int blockM = blockIdx.y * 64;
    const int blockN = blockIdx.x * 64;
    const int waveM  = (wid >> 1) * 16;
    const int waveN  = (wid & 1) * 32;

    v8f acc0 = {}, acc1 = {};

    const int rowA = tid >> 2, kbA = (tid & 3) * 8;
    const int kB   = tid >> 3, nbB = (tid & 7) * 8;

    for (int k0 = 0; k0 < K; k0 += 32) {
        {
            const float4* p = (const float4*)(A + (size_t)(blockM + rowA) * K + k0 + kbA);
            float4 a = p[0], b = p[1];
            PK8 pk;
            pk.h[0] = (_Float16)a.x; pk.h[1] = (_Float16)a.y;
            pk.h[2] = (_Float16)a.z; pk.h[3] = (_Float16)a.w;
            pk.h[4] = (_Float16)b.x; pk.h[5] = (_Float16)b.y;
            pk.h[6] = (_Float16)b.z; pk.h[7] = (_Float16)b.w;
            *(uint4*)&Asub[rowA][kbA] = pk.q;
            if (k0 + 32 < K) __builtin_prefetch((const char*)p + 128, 0, 1);
        }
        {
            const float4* p = (const float4*)(B + (size_t)(k0 + kB) * N + blockN + nbB);
            float4 a = p[0], b = p[1];
            _Float16 h[8] = {(_Float16)a.x, (_Float16)a.y, (_Float16)a.z, (_Float16)a.w,
                             (_Float16)b.x, (_Float16)b.y, (_Float16)b.z, (_Float16)b.w};
            #pragma unroll
            for (int i = 0; i < 8; ++i) Bsub[nbB + i][kB] = h[i];
            if (k0 + 32 < K) __builtin_prefetch((const char*)p + (size_t)32 * N * 4, 0, 1);
        }
        __syncthreads();

        Frag fa, fb0, fb1;
        #pragma unroll
        for (int j = 0; j < 8; ++j) {
            const int ks = kstart(j, g);
            fa.u[j]  = *(const unsigned int*)&Asub[waveM + ln][ks];
            fb0.u[j] = *(const unsigned int*)&Bsub[waveN + ln][ks];
            fb1.u[j] = *(const unsigned int*)&Bsub[waveN + 16 + ln][ks];
        }
        acc0 = wmma_f16(fa, fb0, acc0);
        acc1 = wmma_f16(fa, fb1, acc1);
        __syncthreads();
    }

    const int col0  = blockN + waveN + ln;
    const int col1  = col0 + 16;
    const float bv0 = bias[col0];
    const float bv1 = bias[col1];
    const int rbase = blockM + waveM + 8 * g;
    #pragma unroll
    for (int r = 0; r < 8; ++r) {
        C[(size_t)(rbase + r) * N + col0] = acc0[r] + bv0;
        C[(size_t)(rbase + r) * N + col1] = acc1[r] + bv1;
    }
}

// ---------------------------------------------------------------------------
// Flash attention: 8 waves/block, each wave owns a 16-row Q tile of one head.
// K and V^T chunks (4 KB each) staged by async DMA into padded LDS.
// ---------------------------------------------------------------------------
__global__ __launch_bounds__(256)
void flash_attn_wmma(const _Float16* __restrict__ qkvh,
                     const _Float16* __restrict__ vT,
                     float* __restrict__ attn)
{
    __shared__ _Float16 Kl[32][72];      // [key][dim]  (dim pairs contiguous)
    __shared__ _Float16 Vt[64][40];      // [dim][key]  (key pairs contiguous)
    __shared__ _Float16 Pl[8][16][40];   // per-wave P scratch [row][key]

    const int tid  = threadIdx.x;
    const int lane = tid & 31;
    const int wid  = tid >> 5;
    const int g    = lane >> 4;
    const int ln   = lane & 15;
    const int h    = blockIdx.y;
    const int qbase = (blockIdx.x * 8 + wid) * 16;

    // Q fragments: f16 pair gathers straight from qkvh
    Frag qf[2];
    {
        const _Float16* qrow = qkvh + (size_t)(qbase + ln) * (3 * D_MODEL) + h * DH;
        #pragma unroll
        for (int t = 0; t < 2; ++t)
            #pragma unroll
            for (int j = 0; j < 8; ++j)
                qf[t].u[j] = *(const unsigned int*)(qrow + t * 32 + kstart(j, g));
    }

    v8f oacc[4] = {{}, {}, {}, {}};
    float mrow[8], lsum[8];
    #pragma unroll
    for (int r = 0; r < 8; ++r) { mrow[r] = -1e30f; lsum[r] = 0.f; }

    // async staging slots: K: 32 keys x 64 dims; V^T: 64 dims x 32 keys
    const int kkey = tid >> 3, kdb = (tid & 7) * 8;   // 16B per lane
    const int vdim = tid >> 2, vkq = (tid & 3) * 8;   // 16B per lane
    const float sc = 0.125f;

    for (int c = 0; c < SEQ / 32; ++c) {
        const int kc = c * 32;
        // ---- async DMA K and V^T chunks into LDS (no VALU, ASYNCcnt) ----
        {
            const _Float16* kp = qkvh + (size_t)(kc + kkey) * (3 * D_MODEL)
                                      + D_MODEL + h * DH + kdb;
            const _Float16* vp = vT + (size_t)(h * DH + vdim) * SEQ + kc + vkq;
            unsigned int klds = (unsigned int)(size_t)&Kl[kkey][kdb];
            unsigned int vlds = (unsigned int)(size_t)&Vt[vdim][vkq];
            asm volatile("global_load_async_to_lds_b128 %0, %1, off"
                         :: "v"(klds), "v"((unsigned long long)(size_t)kp) : "memory");
            asm volatile("global_load_async_to_lds_b128 %0, %1, off"
                         :: "v"(vlds), "v"((unsigned long long)(size_t)vp) : "memory");
            if (c + 1 < SEQ / 32) {
                __builtin_prefetch((const char*)kp + (size_t)32 * (3 * D_MODEL) * 2, 0, 1);
                __builtin_prefetch((const char*)vp + 64, 0, 1);
            }
        }
        asm volatile("s_wait_asynccnt 0x0" ::: "memory");
        __syncthreads();

        // ---- S = Q @ K^T : two 16x16 tiles over the 32-key chunk ----
        v8f s0 = {}, s1 = {};
        #pragma unroll
        for (int t = 0; t < 2; ++t) {
            Frag kb0, kb1;
            #pragma unroll
            for (int j = 0; j < 8; ++j) {
                const int ks = t * 32 + kstart(j, g);
                kb0.u[j] = *(const unsigned int*)&Kl[ln][ks];
                kb1.u[j] = *(const unsigned int*)&Kl[16 + ln][ks];
            }
            s0 = wmma_f16(qf[t], kb0, s0);
            s1 = wmma_f16(qf[t], kb1, s1);
        }

        // ---- online softmax ----
        #pragma unroll
        for (int r = 0; r < 8; ++r) {
            float a0 = s0[r] * sc, a1 = s1[r] * sc;
            float mx = fmaxf(a0, a1);
            #pragma unroll
            for (int m = 1; m < 16; m <<= 1)
                mx = fmaxf(mx, __shfl_xor(mx, m, 16));
            const float mnew  = fmaxf(mrow[r], mx);
            const float alpha = __expf(mrow[r] - mnew);
            const float p0 = __expf(a0 - mnew);
            const float p1 = __expf(a1 - mnew);
            float rs = p0 + p1;
            #pragma unroll
            for (int m = 1; m < 16; m <<= 1)
                rs += __shfl_xor(rs, m, 16);
            lsum[r] = lsum[r] * alpha + rs;
            mrow[r] = mnew;
            s0[r] = p0; s1[r] = p1;
            #pragma unroll
            for (int t = 0; t < 4; ++t) oacc[t][r] *= alpha;
        }

        // ---- P: C-layout -> A-fragment via per-wave LDS scratch ----
        #pragma unroll
        for (int r = 0; r < 8; ++r) {
            Pl[wid][r + 8 * g][ln]      = (_Float16)s0[r];
            Pl[wid][r + 8 * g][16 + ln] = (_Float16)s1[r];
        }
        asm volatile("s_wait_dscnt 0x0" ::: "memory");

        Frag pf, vf;
        #pragma unroll
        for (int j = 0; j < 8; ++j)
            pf.u[j] = *(const unsigned int*)&Pl[wid][ln][kstart(j, g)];
        #pragma unroll
        for (int t = 0; t < 4; ++t) {
            #pragma unroll
            for (int j = 0; j < 8; ++j)
                vf.u[j] = *(const unsigned int*)&Vt[t * 16 + ln][kstart(j, g)];
            oacc[t] = wmma_f16(pf, vf, oacc[t]);
        }
        __syncthreads();   // before next chunk's async DMA overwrites Kl/Vt
    }

    float inv[8];
    #pragma unroll
    for (int r = 0; r < 8; ++r) inv[r] = 1.f / lsum[r];
    #pragma unroll
    for (int t = 0; t < 4; ++t) {
        const int col = h * DH + t * 16 + ln;
        #pragma unroll
        for (int r = 0; r < 8; ++r)
            attn[(size_t)(qbase + r + 8 * g) * D_MODEL + col] = oacc[t][r] * inv[r];
    }
}

// ---------------------------------------------------------------------------
extern "C" void kernel_launch(void* const* d_in, const int* in_sizes, int n_in,
                              void* d_out, int out_size, void* d_ws, size_t ws_size,
                              hipStream_t stream)
{
    (void)in_sizes; (void)n_in; (void)out_size; (void)ws_size;
    const float* x     = (const float*)d_in[0];
    const float* w_qkv = (const float*)d_in[1];
    const float* b_qkv = (const float*)d_in[2];
    const float* w_o   = (const float*)d_in[3];
    const float* b_o   = (const float*)d_in[4];
    float* out = (float*)d_out;

    _Float16* qkvh = (_Float16*)d_ws;                               // 4096 x 2304 f16
    _Float16* vT   = qkvh + (size_t)SEQ * 3 * D_MODEL;              // 768 x 4096 f16
    float*    attn = (float*)(vT + (size_t)D_MODEL * SEQ);          // 4096 x 768 f32

    gemm_qkv_wmma<<<dim3((3 * D_MODEL) / 64, SEQ / 64), 256, 0, stream>>>(
        x, w_qkv, b_qkv, qkvh, vT);

    flash_attn_wmma<<<dim3(SEQ / 128, NHEAD), 256, 0, stream>>>(qkvh, vT, attn);

    gemm_bias_wmma<<<dim3(D_MODEL / 64, SEQ / 64), 256, 0, stream>>>(
        attn, w_o, b_o, out, SEQ, D_MODEL, D_MODEL);
}
